// RoPEAttention_88854283419811
// MI455X (gfx1250) — matrix operand order repaired and checked by reading, not compiled, not run on hardware
//
#include <hip/hip_runtime.h>
#include <stdint.h>

// Problem constants (match reference)
constexpr int T_DIM  = 4096;
constexpr int C_DIM  = 2048;
constexpr int H_DIM  = 16;
constexpr int KVH    = 4;
constexpr int D_DIM  = 128;
constexpr int QKD    = H_DIM * D_DIM;   // 2048
constexpr int KVD    = KVH * D_DIM;     // 512

typedef __bf16 bf16_t;
typedef __attribute__((ext_vector_type(16))) __bf16 v16bf;
typedef __attribute__((ext_vector_type(8)))  __bf16 v8bf;
typedef __attribute__((ext_vector_type(8)))  float  v8f;
typedef __attribute__((ext_vector_type(4)))  float  v4f;
typedef unsigned int v4u __attribute__((ext_vector_type(4)));
typedef int v8i __attribute__((ext_vector_type(8)));
typedef int v4i __attribute__((ext_vector_type(4)));

static __device__ __forceinline__ v8f wmma_bf16(v16bf a, v16bf b, v8f c) {
  // D = A(16x32) * B(32x16) + C, f32 accumulate
  return __builtin_amdgcn_wmma_f32_16x16x32_bf16(false, a, false, b, (short)0, c,
                                                 false, false);
}

static __device__ __forceinline__ v8f zero_v8f() {
  v8f z;
#pragma unroll
  for (int i = 0; i < 8; ++i) z[i] = 0.0f;
  return z;
}

static __device__ __forceinline__ v16bf cat16(v8bf lo, v8bf hi) {
  v16bf r;
#pragma unroll
  for (int i = 0; i < 8; ++i) { r[i] = lo[i]; r[i + 8] = hi[i]; }
  return r;
}

// A fragment: 16x32 (MxK), source row-major [16][stride].
// lane m = lane&15, half = lane>>4; elems 0..7 -> K=8*half+i, 8..15 -> K=16+8*half+i
static __device__ __forceinline__ v16bf load_frag_A(const bf16_t* p, int stride) {
  const int lane = threadIdx.x & 31;
  const bf16_t* row = p + (lane & 15) * stride + (lane >> 4) * 8;
  v8bf lo = *(const v8bf*)(row);
  v8bf hi = *(const v8bf*)(row + 16);
  return cat16(lo, hi);
}

// B fragment: 32x16 (KxN), source stored N-major [16][stride] (i.e. B^T row-major).
// lane n = lane&15, half = lane>>4; elems 0..7 -> K=8*half+i, 8..15 -> K=16+8*half+i
static __device__ __forceinline__ v16bf load_frag_B(const bf16_t* p, int stride) {
  const int lane = threadIdx.x & 31;
  const bf16_t* col = p + (lane & 15) * stride + (lane >> 4) * 8;
  v8bf lo = *(const v8bf*)(col);
  v8bf hi = *(const v8bf*)(col + 16);
  return cat16(lo, hi);
}

// ---------------------------------------------------------------------------
// Tensor Data Mover: async 2D bf16 tile (tile_w contiguous elems x tile_h rows,
// row stride in elements) from global into contiguous LDS. Issue from ONE wave
// (TDM ignores EXEC). Tracked with TENSORcnt.
// ---------------------------------------------------------------------------
static __device__ __forceinline__ void tdm_load_2d_bf16(uint32_t lds_addr,
                                                        uint64_t gaddr,
                                                        uint32_t tile_w,
                                                        uint32_t tile_h,
                                                        uint32_t row_stride) {
  v4u g0;
  g0[0] = 1u;                                            // count=1, user mode
  g0[1] = lds_addr;                                      // lds_addr (bytes)
  g0[2] = (uint32_t)gaddr;                               // global_addr[31:0]
  g0[3] = (uint32_t)((gaddr >> 32) & 0x01FFFFFFu) | (2u << 30);  // addr[56:32] | type=2
  v8i g1;
  g1[0] = (int)(1u << 16);                               // data_size=1 -> 2 bytes
  g1[1] = (int)((tile_w & 0xFFFFu) << 16);               // tensor_dim0 lo16
  g1[2] = (int)(((tile_w >> 16) & 0xFFFFu) | (tile_h << 16)); // dim0 hi | tensor_dim1 lo
  g1[3] = (int)((tile_w & 0xFFFFu) << 16);               // tensor_dim1 hi=0 | tile_dim0
  g1[4] = (int)(tile_h & 0xFFFFu);                       // tile_dim1 | tile_dim2=0
  g1[5] = (int)row_stride;                               // tensor_dim0_stride[31:0]
  g1[6] = 0;                                             // stride hi | dim1_stride lo
  g1[7] = 0;
  v4i z = {0, 0, 0, 0};
#if defined(__clang_major__) && (__clang_major__ >= 23)
  v8i z8 = {0, 0, 0, 0, 0, 0, 0, 0};
  __builtin_amdgcn_tensor_load_to_lds(g0, g1, z, z, z8, 0);
#else
  __builtin_amdgcn_tensor_load_to_lds(g0, g1, z, z, 0);
#endif
}

static __device__ __forceinline__ uint32_t lds_off(const void* p) {
  return (uint32_t)(uintptr_t)p;   // LDS aperture: low 32 bits = LDS byte address
}

// ---------------------------------------------------------------------------
// Kernel 1: Y[T][NOUT] = bf16( rope?( X[T][C] @ W[C][NOUT] ) )
// Block: 256 thr (8 waves), tile 128(M) x 64(N), K-step 32.
// ---------------------------------------------------------------------------
template <int NOUT, bool ROPE>
__global__ __launch_bounds__(256)
void qkv_proj_rope_kernel(const float* __restrict__ X, const float* __restrict__ W,
                          bf16_t* __restrict__ Y) {
  __shared__ __align__(16) bf16_t sA[128 * 32];   // [M][K]
  __shared__ __align__(16) bf16_t sB[64 * 32];    // [N][K]  (transposed weight tile)
  const int tid = threadIdx.x;
  const int w = tid >> 5, lane = tid & 31;
  const int nl = lane & 15, hl = lane >> 4;
  const int wm = w & 3, wn = w >> 2;              // 4 x 2 wave grid
  const int m0 = blockIdx.y * 128, n0 = blockIdx.x * 64;

  v8f acc[2][2];
#pragma unroll
  for (int i = 0; i < 2; ++i)
#pragma unroll
    for (int j = 0; j < 2; ++j) acc[i][j] = zero_v8f();

  for (int k0 = 0; k0 < C_DIM; k0 += 32) {
    __syncthreads();
    // Stage A: 128x32 f32 -> bf16 (1024 float4 slots, 4 per thread)
#pragma unroll
    for (int it = 0; it < 4; ++it) {
      int slot = it * 256 + tid;
      int r = slot >> 3, c4 = (slot & 7) * 4;
      v4f v = *(const v4f*)(X + (size_t)(m0 + r) * C_DIM + k0 + c4);
      bf16_t* d = sA + r * 32 + c4;
      d[0] = (bf16_t)v[0]; d[1] = (bf16_t)v[1];
      d[2] = (bf16_t)v[2]; d[3] = (bf16_t)v[3];
    }
    // Stage B transposed: W[k][n] -> sB[n*32+k] (512 float4 slots, 2 per thread)
#pragma unroll
    for (int it = 0; it < 2; ++it) {
      int slot = it * 256 + tid;
      int kk = slot >> 4, n4 = (slot & 15) * 4;
      v4f v = *(const v4f*)(W + (size_t)(k0 + kk) * NOUT + n0 + n4);
      sB[(n4 + 0) * 32 + kk] = (bf16_t)v[0];
      sB[(n4 + 1) * 32 + kk] = (bf16_t)v[1];
      sB[(n4 + 2) * 32 + kk] = (bf16_t)v[2];
      sB[(n4 + 3) * 32 + kk] = (bf16_t)v[3];
    }
    __syncthreads();

    const bf16_t* Ab = sA + (wm * 32) * 32;
    const bf16_t* Bb = sB + (wn * 32) * 32;
    v16bf a0 = load_frag_A(Ab, 32);
    v16bf a1 = load_frag_A(Ab + 16 * 32, 32);
    v16bf b0 = load_frag_B(Bb, 32);
    v16bf b1 = load_frag_B(Bb + 16 * 32, 32);
    acc[0][0] = wmma_bf16(a0, b0, acc[0][0]);
    acc[0][1] = wmma_bf16(a0, b1, acc[0][1]);
    acc[1][0] = wmma_bf16(a1, b0, acc[1][0]);
    acc[1][1] = wmma_bf16(a1, b1, acc[1][1]);
  }

  // Epilogue: optional RoPE (even/odd head-dims sit in adjacent lanes), store bf16
#pragma unroll
  for (int mi_ = 0; mi_ < 2; ++mi_)
#pragma unroll
    for (int ni = 0; ni < 2; ++ni)
#pragma unroll
      for (int r = 0; r < 8; ++r) {
        float val = acc[mi_][ni][r];
        int row = m0 + wm * 32 + mi_ * 16 + r + 8 * hl;
        int col = n0 + wn * 32 + ni * 16 + nl;
        if (ROPE) {
          float partner = __shfl_xor(val, 1, 32);
          int dd = col & (D_DIM - 1);
          // inv = theta^(-(2i)/D) = exp(-(dd&~1) * ln(theta)/128)
          float inv = __expf(-(float)(dd & ~1) * (9.210340371976184f / 128.0f));
          float ang = (float)row * inv;
          float cs, sn; __sincosf(ang, &cs, &sn);
          val = (dd & 1) ? (partner * sn + val * cs) : (val * cs - partner * sn);
        }
        Y[(size_t)row * NOUT + col] = (bf16_t)val;
      }
}

// ---------------------------------------------------------------------------
// Kernel 2: causal GQA flash attention. grid=(T/128, H), 256 thr (8 waves).
// Each wave: 16 q rows x full D=128. Key tiles of 32 streamed through LDS;
// K tile arrives via the Tensor Data Mover (async DMA, TENSORcnt).
// B-fragment loads are software-pipelined in pairs so DS waits stagger
// instead of fully serializing each WMMA behind its own loads.
// ---------------------------------------------------------------------------
__global__ __launch_bounds__(256)
void flash_attn_kernel(const bf16_t* __restrict__ Q, const bf16_t* __restrict__ K,
                       const bf16_t* __restrict__ V, bf16_t* __restrict__ O) {
  __shared__ __align__(16) bf16_t sK[32 * 128];    // [kj][d]   (B-frag ready for S=QK^T)
  __shared__ __align__(16) bf16_t sVt[128 * 32];   // [d][kj]   (B-frag ready for P*V)
  __shared__ __align__(16) bf16_t sP[8][16 * 32];  // per-wave P staging
  const int h = blockIdx.y;
  const int kvh = h >> 2;                          // GQA: 4 q-heads per kv-head
  const int qb = blockIdx.x;
  const int tid = threadIdx.x, w = tid >> 5, lane = tid & 31;
  const int nl = lane & 15, hl = lane >> 4;
  const int q0 = qb * 128;
  const int qrow0 = q0 + w * 16;

  // Preload this wave's Q fragments (4 K-chunks of 32 over D=128)
  v16bf qf[4];
#pragma unroll
  for (int dc = 0; dc < 4; ++dc)
    qf[dc] = load_frag_A(Q + (size_t)qrow0 * QKD + h * D_DIM + dc * 32, QKD);

  v8f o[8];
#pragma unroll
  for (int j = 0; j < 8; ++j) o[j] = zero_v8f();
  float mi[8], li[8];
#pragma unroll
  for (int r = 0; r < 8; ++r) { mi[r] = -__builtin_inff(); li[r] = 0.0f; }

  const float scale = 0.08838834764831845f;        // 1/sqrt(128)
  const int ktiles = (qb + 1) * 4;                 // causal: keys <= q0+127

  for (int kt = 0; kt < ktiles; ++kt) {
    const int k0 = kt * 32;
    __syncthreads();
    // Async DMA of K tile (32 rows x 128 d, row stride KVD) into LDS via TDM.
    if (w == 0) {
      tdm_load_2d_bf16(lds_off(sK),
                       (uint64_t)(uintptr_t)(K + (size_t)k0 * KVD + kvh * D_DIM),
                       /*tile_w=*/D_DIM, /*tile_h=*/32, /*row_stride=*/KVD);
    }
    // Meanwhile: stage V tile transposed with the VALU (scalar transpose writes)
#pragma unroll
    for (int it = 0; it < 16; ++it) {
      int idx = it * 256 + tid;
      int r = idx >> 7, dd = idx & 127;
      sVt[dd * 32 + r] = V[(size_t)(k0 + r) * KVD + kvh * D_DIM + dd];
    }
    if (w == 0) __builtin_amdgcn_s_wait_tensorcnt(0);
    __syncthreads();

    // S = Q @ K^T  (two 16x16 n-subtiles over the 32 keys).
    // Load fragments for two K-chunks ahead of their WMMAs (pipelined pairs).
    v8f s0 = zero_v8f(), s1 = zero_v8f();
#pragma unroll
    for (int dc = 0; dc < 4; dc += 2) {
      v16bf b00 = load_frag_B(sK + dc * 32, 128);
      v16bf b01 = load_frag_B(sK + 16 * 128 + dc * 32, 128);
      v16bf b10 = load_frag_B(sK + (dc + 1) * 32, 128);
      v16bf b11 = load_frag_B(sK + 16 * 128 + (dc + 1) * 32, 128);
      s0 = wmma_bf16(qf[dc], b00, s0);
      s1 = wmma_bf16(qf[dc], b01, s1);
      s0 = wmma_bf16(qf[dc + 1], b10, s0);
      s1 = wmma_bf16(qf[dc + 1], b11, s1);
    }

    // Online softmax per row (rows live in C-layout: m = r + 8*hl, cols across 16 lanes)
#pragma unroll
    for (int r = 0; r < 8; ++r) {
      const int qrow = qrow0 + r + 8 * hl;
      float v0 = s0[r] * scale;
      float v1 = s1[r] * scale;
      if (k0 + nl > qrow)      v0 = -__builtin_inff();
      if (k0 + 16 + nl > qrow) v1 = -__builtin_inff();
      float mx = fmaxf(v0, v1);
#pragma unroll
      for (int off = 1; off < 16; off <<= 1) mx = fmaxf(mx, __shfl_xor(mx, off, 32));
      float mnew = fmaxf(mi[r], mx);
      float alpha = __expf(mi[r] - mnew);          // 0 on first tile
      mi[r] = mnew;
      float p0 = __expf(v0 - mnew);
      float p1 = __expf(v1 - mnew);
      float rs = p0 + p1;
#pragma unroll
      for (int off = 1; off < 16; off <<= 1) rs += __shfl_xor(rs, off, 32);
      li[r] = li[r] * alpha + rs;
#pragma unroll
      for (int j = 0; j < 8; ++j) o[j][r] *= alpha;
      sP[w][(r + 8 * hl) * 32 + nl]      = (bf16_t)p0;
      sP[w][(r + 8 * hl) * 32 + 16 + nl] = (bf16_t)p1;
    }
    // Same-wave LDS RAW: wait on DS counter before reading P back
    asm volatile("s_wait_dscnt 0x0" ::: "memory");

    // O += P @ V  (one A-frag covers all 32 keys; 8 output d-subtiles, pipelined pairs)
    v16bf pf = load_frag_A(&sP[w][0], 32);
#pragma unroll
    for (int j = 0; j < 8; j += 2) {
      v16bf bv0 = load_frag_B(sVt + (j * 16) * 32, 32);
      v16bf bv1 = load_frag_B(sVt + ((j + 1) * 16) * 32, 32);
      o[j]     = wmma_bf16(pf, bv0, o[j]);
      o[j + 1] = wmma_bf16(pf, bv1, o[j + 1]);
    }
  }

  // Normalize and store attention output (bf16, [T][H*D])
#pragma unroll
  for (int j = 0; j < 8; ++j)
#pragma unroll
    for (int r = 0; r < 8; ++r) {
      float val = o[j][r] / li[r];
      int row = qrow0 + r + 8 * hl;
      int col = h * D_DIM + j * 16 + nl;
      O[(size_t)row * QKD + col] = (bf16_t)val;
    }
}

// ---------------------------------------------------------------------------
// Kernel 3: OUT[T][C] = f32( A_bf16[T][2048] @ Wo[2048][2048] )
// A tile arrives via TDM while the VALU converts the f32 weight tile.
// ---------------------------------------------------------------------------
__global__ __launch_bounds__(256)
void out_proj_kernel(const bf16_t* __restrict__ A, const float* __restrict__ W,
                     float* __restrict__ OUT) {
  __shared__ __align__(16) bf16_t sA[128 * 32];
  __shared__ __align__(16) bf16_t sB[64 * 32];
  const int tid = threadIdx.x;
  const int w = tid >> 5, lane = tid & 31;
  const int nl = lane & 15, hl = lane >> 4;
  const int wm = w & 3, wn = w >> 2;
  const int m0 = blockIdx.y * 128, n0 = blockIdx.x * 64;

  v8f acc[2][2];
#pragma unroll
  for (int i = 0; i < 2; ++i)
#pragma unroll
    for (int j = 0; j < 2; ++j) acc[i][j] = zero_v8f();

  for (int k0 = 0; k0 < QKD; k0 += 32) {
    __syncthreads();
    // Async DMA of A tile (128 rows x 32 elems, row stride QKD) into LDS via TDM.
    if (w == 0) {
      tdm_load_2d_bf16(lds_off(sA),
                       (uint64_t)(uintptr_t)(A + (size_t)m0 * QKD + k0),
                       /*tile_w=*/32, /*tile_h=*/128, /*row_stride=*/QKD);
    }
    // Meanwhile: stage B transposed from f32 Wo with the VALU
#pragma unroll
    for (int it = 0; it < 2; ++it) {
      int slot = it * 256 + tid;
      int kk = slot >> 4, n4 = (slot & 15) * 4;
      v4f v = *(const v4f*)(W + (size_t)(k0 + kk) * C_DIM + n0 + n4);
      sB[(n4 + 0) * 32 + kk] = (bf16_t)v[0];
      sB[(n4 + 1) * 32 + kk] = (bf16_t)v[1];
      sB[(n4 + 2) * 32 + kk] = (bf16_t)v[2];
      sB[(n4 + 3) * 32 + kk] = (bf16_t)v[3];
    }
    if (w == 0) __builtin_amdgcn_s_wait_tensorcnt(0);
    __syncthreads();

    const bf16_t* Ab = sA + (wm * 32) * 32;
    const bf16_t* Bb = sB + (wn * 32) * 32;
    v16bf a0 = load_frag_A(Ab, 32);
    v16bf a1 = load_frag_A(Ab + 16 * 32, 32);
    v16bf b0 = load_frag_B(Bb, 32);
    v16bf b1 = load_frag_B(Bb + 16 * 32, 32);
    acc[0][0] = wmma_bf16(a0, b0, acc[0][0]);
    acc[0][1] = wmma_bf16(a0, b1, acc[0][1]);
    acc[1][0] = wmma_bf16(a1, b0, acc[1][0]);
    acc[1][1] = wmma_bf16(a1, b1, acc[1][1]);
  }

#pragma unroll
  for (int mi_ = 0; mi_ < 2; ++mi_)
#pragma unroll
    for (int ni = 0; ni < 2; ++ni)
#pragma unroll
      for (int r = 0; r < 8; ++r) {
        int row = m0 + wm * 32 + mi_ * 16 + r + 8 * hl;
        int col = n0 + wn * 32 + ni * 16 + nl;
        OUT[(size_t)row * C_DIM + col] = acc[mi_][ni][r];
      }
}

// ---------------------------------------------------------------------------
extern "C" void kernel_launch(void* const* d_in, const int* in_sizes, int n_in,
                              void* d_out, int out_size, void* d_ws, size_t ws_size,
                              hipStream_t stream) {
  (void)in_sizes; (void)n_in; (void)out_size; (void)ws_size;
  const float* x  = (const float*)d_in[0];
  // d_in[1] = causal mask (implemented analytically; unused)
  const float* Wq = (const float*)d_in[2];
  const float* Wk = (const float*)d_in[3];
  const float* Wv = (const float*)d_in[4];
  const float* Wo = (const float*)d_in[5];
  float* out = (float*)d_out;

  char* ws = (char*)d_ws;
  const size_t QB_BYTES = (size_t)T_DIM * QKD * sizeof(bf16_t);  // 16 MB
  const size_t KB_BYTES = (size_t)T_DIM * KVD * sizeof(bf16_t);  //  4 MB
  bf16_t* qbuf = (bf16_t*)(ws);
  bf16_t* kbuf = (bf16_t*)(ws + QB_BYTES);
  bf16_t* vbuf = (bf16_t*)(ws + QB_BYTES + KB_BYTES);
  bf16_t* abuf = (bf16_t*)(ws + QB_BYTES + 2 * KB_BYTES);

  // QKV projections (+RoPE on q,k)
  qkv_proj_rope_kernel<QKD, true><<<dim3(QKD / 64, T_DIM / 128), 256, 0, stream>>>(x, Wq, qbuf);
  qkv_proj_rope_kernel<KVD, true><<<dim3(KVD / 64, T_DIM / 128), 256, 0, stream>>>(x, Wk, kbuf);
  qkv_proj_rope_kernel<KVD, false><<<dim3(KVD / 64, T_DIM / 128), 256, 0, stream>>>(x, Wv, vbuf);
  // Flash attention
  flash_attn_kernel<<<dim3(T_DIM / 128, H_DIM), 256, 0, stream>>>(qbuf, kbuf, vbuf, abuf);
  // Output projection
  out_proj_kernel<<<dim3(C_DIM / 64, T_DIM / 128), 256, 0, stream>>>(abuf, Wo, out);
}
